// FastDctRNN_85899345920339
// MI455X (gfx1250) — matrix operand order repaired and checked by reading, not compiled, not run on hardware
//
#include <hip/hip_runtime.h>

// ---------------------------------------------------------------------------
// FastDctRNN for MI455X (gfx1250, wave32, WMMA)
//
// Dims: I=256, H=512, F=3831 (c_ih=1275, c_hh=2556), 4F=15324.
// Padded: F_PAD=3840, NG_PAD=15328 (958 x 16 N-tiles = 479 N-pairs).
//
// Gates GEMM: per-wave tile = 4 M-tiles x 2 N-tiles (8 accумulators), so each
// 32-K step loads 4 A-frags + 2 B-frags for 8 WMMAs (0.75 loads/WMMA; A
// traffic halved vs 1-N-tile).  Unified K space (256 ih + 3840 hh = 4096),
// split-K x8 -> 479*8 = 3832 waves.  K-loop unrolled x2 with ping-pong
// fragment sets so loads stay in flight across the WMMA groups.
// ---------------------------------------------------------------------------

typedef __attribute__((ext_vector_type(16))) __bf16 v16bf;
typedef __attribute__((ext_vector_type(8)))  __bf16 v8bf;
typedef __attribute__((ext_vector_type(8)))  float  v8f;

#define SEQ_T   64
#define BATCH_B 64
#define I_DIM   256
#define H_DIM   512
#define F_DIM   3831
#define F_PAD   3840
#define NG      (4 * F_DIM)      // 15324
#define NG_PAD  15328
#define N_TILES (NG_PAD / 16)    // 958
#define N_PAIRS (N_TILES / 2)    // 479
#define KSPLIT  8                // split-K factor (4096 / 512)
#define KCHUNK  512
#define PI_F    3.14159265358979323846f

__device__ __forceinline__ __bf16 f2bf(float f) {
  union { float f; unsigned u; } v; v.f = f;
  unsigned r = v.u + 0x7FFFu + ((v.u >> 16) & 1u);   // round-to-nearest-even
  unsigned short h = (unsigned short)(r >> 16);
  return __builtin_bit_cast(__bf16, h);
}

__device__ __forceinline__ float sigmoidf_(float x) {
  return 1.0f / (1.0f + __expf(-x));
}

// ---------------- one-time prep kernels ----------------

__global__ void k_conv_wih(const float* __restrict__ w, __bf16* __restrict__ o) {
  int idx = blockIdx.x * blockDim.x + threadIdx.x;
  if (idx >= NG_PAD * I_DIM) return;
  int j = idx / I_DIM, i = idx - j * I_DIM;
  o[idx] = (j < NG) ? f2bf(w[j * I_DIM + i]) : f2bf(0.0f);
}

__global__ void k_conv_whh(const float* __restrict__ w, __bf16* __restrict__ o) {
  long idx = (long)blockIdx.x * blockDim.x + threadIdx.x;
  if (idx >= (long)NG_PAD * F_PAD) return;
  int j = (int)(idx / F_PAD), k = (int)(idx - (long)j * F_PAD);
  o[idx] = (j < NG && k < F_DIM) ? f2bf(w[(long)j * F_DIM + k]) : f2bf(0.0f);
}

__global__ void k_dct(float* __restrict__ D, int N) {
  int idx = blockIdx.x * blockDim.x + threadIdx.x;
  if (idx >= N * N) return;
  int k = idx / N, n = idx - k * N;
  float s = sqrtf(2.0f / (float)N) * ((k == 0) ? sqrtf(0.5f) : 1.0f);
  D[idx] = cosf(PI_F * (2.0f * (float)n + 1.0f) * (float)k / (2.0f * (float)N)) * s;
}

__global__ void k_conv_x(const float* __restrict__ x, __bf16* __restrict__ o) {
  int idx = blockIdx.x * blockDim.x + threadIdx.x;
  if (idx >= BATCH_B * I_DIM) return;
  o[idx] = f2bf(x[idx]);
}

// ---------------- WMMA fragment loaders (CDNA5 wave32 layouts) ----------------

// A (16x32 bf16, row-major, stride lda): lanes 0-15 -> M=lane, K in {k0..+7, k0+16..+23}
//                                        lanes 16-31 -> M=lane-16, K in {+8..+15, +24..+31}
__device__ __forceinline__ v16bf load_a_frag(const __bf16* __restrict__ A, int lda,
                                             int m_base, int k0, int lane) {
  int l = lane & 15, hi = lane >> 4;
  const __bf16* p = A + (m_base + l) * lda + k0 + hi * 8;
  v8bf lo = *(const v8bf*)(p);
  v8bf hh = *(const v8bf*)(p + 16);
  v16bf a;
#pragma unroll
  for (int i = 0; i < 8; ++i) { a[i] = lo[i]; a[8 + i] = hh[i]; }
  return a;
}

// B (32x16 bf16) from row-major W (N x K, stride ldw), B[k,n] = W[n,k]:
// lanes 0-15 hold K in {k0..+7, k0+16..+23} for col n=lane; lanes 16-31 hold K in {+8..+15, +24..+31}.
__device__ __forceinline__ v16bf load_b_frag(const __bf16* __restrict__ W, int ldw,
                                             int n_base, int k0, int lane) {
  int l = lane & 15, koff = (lane >> 4) * 8;
  const __bf16* p = W + (n_base + l) * ldw + k0 + koff;
  v8bf lo = *(const v8bf*)(p);
  v8bf hh = *(const v8bf*)(p + 16);
  v16bf b;
#pragma unroll
  for (int i = 0; i < 8; ++i) { b[i] = lo[i]; b[8 + i] = hh[i]; }
  return b;
}

#define WMMA_BF16(a, b, c) \
  __builtin_amdgcn_wmma_f32_16x16x32_bf16(false, (a), false, (b), (short)0, (c), false, false)

// One GEMM segment over a 4Mx2N tile.  (kend - kbeg) must be a positive
// multiple of 64.  K-loop unrolled x2, ping-pong fragment sets; trailing
// prefetch over-reads <=64 elements past the segment end (buffers padded).
__device__ __forceinline__ void gemm_seg(const __bf16* __restrict__ A, int lda,
                                         const __bf16* __restrict__ W, int ldw,
                                         int nb0, int kbeg, int kend,
                                         int lane, v8f acc[8]) {
  v16bf b00 = load_b_frag(W, ldw, nb0,      kbeg, lane);
  v16bf b01 = load_b_frag(W, ldw, nb0 + 16, kbeg, lane);
  v16bf a00 = load_a_frag(A, lda,  0, kbeg, lane);
  v16bf a01 = load_a_frag(A, lda, 16, kbeg, lane);
  v16bf a02 = load_a_frag(A, lda, 32, kbeg, lane);
  v16bf a03 = load_a_frag(A, lda, 48, kbeg, lane);
#pragma unroll 1
  for (int k0 = kbeg; k0 < kend; k0 += 64) {
    // prefetch set1 (k0+32)
    v16bf b10 = load_b_frag(W, ldw, nb0,      k0 + 32, lane);
    v16bf b11 = load_b_frag(W, ldw, nb0 + 16, k0 + 32, lane);
    v16bf a10 = load_a_frag(A, lda,  0, k0 + 32, lane);
    v16bf a11 = load_a_frag(A, lda, 16, k0 + 32, lane);
    v16bf a12 = load_a_frag(A, lda, 32, k0 + 32, lane);
    v16bf a13 = load_a_frag(A, lda, 48, k0 + 32, lane);
    // consume set0 (k0)
    acc[0] = WMMA_BF16(a00, b00, acc[0]);
    acc[1] = WMMA_BF16(a01, b00, acc[1]);
    acc[2] = WMMA_BF16(a02, b00, acc[2]);
    acc[3] = WMMA_BF16(a03, b00, acc[3]);
    acc[4] = WMMA_BF16(a00, b01, acc[4]);
    acc[5] = WMMA_BF16(a01, b01, acc[5]);
    acc[6] = WMMA_BF16(a02, b01, acc[6]);
    acc[7] = WMMA_BF16(a03, b01, acc[7]);
    // prefetch set0 (k0+64) -- unconditional, may over-read past kend
    b00 = load_b_frag(W, ldw, nb0,      k0 + 64, lane);
    b01 = load_b_frag(W, ldw, nb0 + 16, k0 + 64, lane);
    a00 = load_a_frag(A, lda,  0, k0 + 64, lane);
    a01 = load_a_frag(A, lda, 16, k0 + 64, lane);
    a02 = load_a_frag(A, lda, 32, k0 + 64, lane);
    a03 = load_a_frag(A, lda, 48, k0 + 64, lane);
    // consume set1 (k0+32)
    acc[0] = WMMA_BF16(a10, b10, acc[0]);
    acc[1] = WMMA_BF16(a11, b10, acc[1]);
    acc[2] = WMMA_BF16(a12, b10, acc[2]);
    acc[3] = WMMA_BF16(a13, b10, acc[3]);
    acc[4] = WMMA_BF16(a10, b11, acc[4]);
    acc[5] = WMMA_BF16(a11, b11, acc[5]);
    acc[6] = WMMA_BF16(a12, b11, acc[6]);
    acc[7] = WMMA_BF16(a13, b11, acc[7]);
  }
}

// ---------------- main gates GEMM, 4Mx2N tiles, split-K x8 ----------------
// Unified K space: [0,256) = ih, [256,4096) = hh.  Chunk c covers
// [c*512, (c+1)*512).  One wave per (n-pair, chunk); 479*8 = 3832 waves.
__global__ void __launch_bounds__(128)
k_gemm_gates(const __bf16* __restrict__ X,     // 64 x 256
             const __bf16* __restrict__ Hl,    // 64 x 3840
             const __bf16* __restrict__ Wih,   // 15328 x 256
             const __bf16* __restrict__ Whh,   // 15328 x 3840
             float* __restrict__ part)         // KSPLIT x 64 x 15328
{
  int lane = threadIdx.x;                 // 0..31
  int np = blockIdx.x * 4 + threadIdx.y;  // n-pair
  if (np >= N_PAIRS) return;              // wave-uniform
  int chunk = blockIdx.y;                 // 0..KSPLIT-1
  int nb0 = np * 32;

  v8f acc[8] = {v8f{}, v8f{}, v8f{}, v8f{}, v8f{}, v8f{}, v8f{}, v8f{}};

  if (chunk == 0) {
    gemm_seg(X,  I_DIM, Wih, I_DIM, nb0, 0, I_DIM, lane, acc);           // ih: K 0..256
    gemm_seg(Hl, F_PAD, Whh, F_PAD, nb0, 0, KCHUNK - I_DIM, lane, acc);  // hh: K 0..256
  } else {
    int hb = (KCHUNK - I_DIM) + (chunk - 1) * KCHUNK;                    // 256 + ...
    gemm_seg(Hl, F_PAD, Whh, F_PAD, nb0, hb, hb + KCHUNK, lane, acc);
  }

  // C/D layout -> row = mt*16 + 8*(lane>=16) + r, col = n_base + (lane&15)
  float* dst = part + (size_t)chunk * BATCH_B * NG_PAD;
  int rbase = (lane >> 4) * 8;
#pragma unroll
  for (int nt = 0; nt < 2; ++nt) {
    int col = nb0 + nt * 16 + (lane & 15);
#pragma unroll
    for (int mt = 0; mt < 4; ++mt) {
#pragma unroll
      for (int r = 0; r < 8; ++r) {
        int row = mt * 16 + rbase + r;
        dst[(long)row * NG_PAD + col] = acc[nt * 4 + mt][r];
      }
    }
  }
}

// ---------------- reduce split-K partials + biases ----------------
__global__ void k_reduce_gates(const float* __restrict__ part,
                               const float* __restrict__ b_ih,
                               const float* __restrict__ b_hh,
                               float* __restrict__ gates) {
  long idx = (long)blockIdx.x * blockDim.x + threadIdx.x;
  if (idx >= (long)BATCH_B * NG_PAD) return;
  int col = (int)(idx % NG_PAD);
  float s = part[idx];
#pragma unroll
  for (int c = 1; c < KSPLIT; ++c) s += part[(size_t)c * BATCH_B * NG_PAD + idx];
  if (col < NG) s += b_ih[col] + b_hh[col];
  gates[idx] = s;
}

// ---------------- LSTM activations ----------------
__global__ void k_lstm(const float* __restrict__ gates, float* __restrict__ c_l,
                       float* __restrict__ h_l, __bf16* __restrict__ h_l_bf) {
  int idx = blockIdx.x * blockDim.x + threadIdx.x;
  if (idx >= BATCH_B * F_DIM) return;
  int b = idx / F_DIM, k = idx - b * F_DIM;
  const float* g = gates + (long)b * NG_PAD;
  float ig = sigmoidf_(g[k]);
  float fg = sigmoidf_(g[F_DIM + k]);
  float gg = tanhf(g[2 * F_DIM + k]);
  float og = sigmoidf_(g[3 * F_DIM + k]);
  float c  = fg * c_l[idx] + ig * gg;
  float h  = og * tanhf(c);
  c_l[idx] = c;
  h_l[idx] = h;
  h_l_bf[b * F_PAD + k] = f2bf(h);   // pad cols stay zero from init
}

// ---------------- forward DCT: u[b,m] = sum_n D[m,n] * xin[b,n] ----------------
__global__ void k_dct_fwd(const float* __restrict__ D, const float* __restrict__ xin,
                          float* __restrict__ u, int N) {
  int idx = blockIdx.x * blockDim.x + threadIdx.x;
  if (idx >= BATCH_B * N) return;
  int b = idx / N, m = idx - b * N;
  const float* xr = xin + b * N;
  const float* Dr = D + m * N;
  float acc = 0.0f;
#pragma unroll 4
  for (int n = 0; n < N; ++n) acc = fmaf(Dr[n], xr[n], acc);
  u[idx] = acc;
}

// ---------------- sparse scatter with LDS float atomics ----------------
__global__ void k_sparse(const float* __restrict__ h_l, const float* __restrict__ scaler,
                         const int* __restrict__ ih_r, const int* __restrict__ ih_c, int c_ih,
                         const int* __restrict__ hh_r, const int* __restrict__ hh_c, int c_hh,
                         const float* __restrict__ u_ih, const float* __restrict__ u_hh,
                         float* __restrict__ v) {
  __shared__ float vs[H_DIM];
  int b = blockIdx.x;
  for (int k = threadIdx.x; k < H_DIM; k += blockDim.x) vs[k] = 0.0f;
  __syncthreads();
  const float* hl = h_l + (long)b * F_DIM;
  const float* ui = u_ih + b * I_DIM;
  const float* uh = u_hh + b * H_DIM;
  for (int e = threadIdx.x; e < c_ih; e += blockDim.x) {
    float coeff = scaler[e] * hl[e];
    atomicAdd(&vs[ih_r[e]], coeff * ui[(I_DIM - 1) - ih_c[e]]);   // col-reversal folded in
  }
  for (int e = threadIdx.x; e < c_hh; e += blockDim.x) {
    int f = c_ih + e;
    float coeff = scaler[f] * hl[f];
    atomicAdd(&vs[hh_r[e]], coeff * uh[(H_DIM - 1) - hh_c[e]]);
  }
  __syncthreads();
  for (int k = threadIdx.x; k < H_DIM; k += blockDim.x) v[b * H_DIM + k] = vs[k];
}

// ---------------- final: h_new = tanh(D_hid^T @ v + bias) ----------------
__global__ void k_final(const float* __restrict__ Dhid, const float* __restrict__ v,
                        const float* __restrict__ bias, float* __restrict__ h,
                        float* __restrict__ out_t) {
  int idx = blockIdx.x * blockDim.x + threadIdx.x;
  if (idx >= BATCH_B * H_DIM) return;
  int b = idx / H_DIM, m = idx - b * H_DIM;
  const float* vr = v + b * H_DIM;
  float acc = bias[m];
#pragma unroll 4
  for (int k = 0; k < H_DIM; ++k) acc = fmaf(Dhid[k * H_DIM + m], vr[k], acc);
  float hn = tanhf(acc);
  h[idx] = hn;
  out_t[idx] = hn;
}

// ---------------------------------------------------------------------------

extern "C" void kernel_launch(void* const* d_in, const int* in_sizes, int n_in,
                              void* d_out, int out_size, void* d_ws, size_t ws_size,
                              hipStream_t stream) {
  const float* input_ = (const float*)d_in[0];
  const float* w_ih   = (const float*)d_in[1];
  const float* w_hh   = (const float*)d_in[2];
  const float* b_ih   = (const float*)d_in[3];
  const float* b_hh   = (const float*)d_in[4];
  const float* scaler = (const float*)d_in[5];
  const float* bias   = (const float*)d_in[6];
  const int* ih_r = (const int*)d_in[7];
  const int* ih_c = (const int*)d_in[8];
  const int* hh_r = (const int*)d_in[9];
  const int* hh_c = (const int*)d_in[10];
  const int c_ih = in_sizes[7];
  const int c_hh = in_sizes[9];

  // carve workspace (256B aligned); staging buffers padded +512B for the
  // unconditional prefetch over-read.  Total ~165 MB.
  char* ws = (char*)d_ws;
  size_t off = 0;
  auto carve = [&](size_t bytes) -> void* {
    void* p = ws + off;
    off = (off + bytes + 255) & ~(size_t)255;
    return p;
  };
  __bf16* wih_bf = (__bf16*)carve((size_t)NG_PAD * I_DIM * 2 + 512);
  __bf16* whh_bf = (__bf16*)carve((size_t)NG_PAD * F_PAD * 2 + 512);
  float*  Din    = (float*)carve((size_t)I_DIM * I_DIM * 4);
  float*  Dhid   = (float*)carve((size_t)H_DIM * H_DIM * 4);
  __bf16* x_bf   = (__bf16*)carve((size_t)BATCH_B * I_DIM * 2 + 512);
  float*  gpart  = (float*)carve((size_t)KSPLIT * BATCH_B * NG_PAD * 4);
  float*  gates  = (float*)carve((size_t)BATCH_B * NG_PAD * 4);
  float*  c_l    = (float*)carve((size_t)BATCH_B * F_DIM * 4);
  float*  h_l    = (float*)carve((size_t)BATCH_B * F_DIM * 4);
  __bf16* h_l_bf = (__bf16*)carve((size_t)BATCH_B * F_PAD * 2 + 512);
  float*  h      = (float*)carve((size_t)BATCH_B * H_DIM * 4);
  float*  u_ih   = (float*)carve((size_t)BATCH_B * I_DIM * 4);
  float*  u_hh   = (float*)carve((size_t)BATCH_B * H_DIM * 4);
  float*  v      = (float*)carve((size_t)BATCH_B * H_DIM * 4);

  // deterministic per-call init (capture-safe memset nodes)
  hipMemsetAsync(c_l,    0, (size_t)BATCH_B * F_DIM * 4, stream);
  hipMemsetAsync(h_l,    0, (size_t)BATCH_B * F_DIM * 4, stream);
  hipMemsetAsync(h_l_bf, 0, (size_t)BATCH_B * F_PAD * 2, stream);
  hipMemsetAsync(h,      0, (size_t)BATCH_B * H_DIM * 4, stream);

  // one-time prep
  k_conv_wih<<<(NG_PAD * I_DIM + 255) / 256, 256, 0, stream>>>(w_ih, wih_bf);
  {
    long tot = (long)NG_PAD * F_PAD;
    k_conv_whh<<<(unsigned)((tot + 255) / 256), 256, 0, stream>>>(w_hh, whh_bf);
  }
  k_dct<<<(I_DIM * I_DIM + 255) / 256, 256, 0, stream>>>(Din, I_DIM);
  k_dct<<<(H_DIM * H_DIM + 255) / 256, 256, 0, stream>>>(Dhid, H_DIM);

  const long ngelem = (long)BATCH_B * NG_PAD;

  // sequential LSTM scan
  for (int t = 0; t < SEQ_T; ++t) {
    const float* x_t = input_ + (size_t)t * BATCH_B * I_DIM;
    k_conv_x<<<(BATCH_B * I_DIM + 255) / 256, 256, 0, stream>>>(x_t, x_bf);

    dim3 gblk(32, 4);                        // 4 waves/block, one n-pair each
    dim3 ggrd((N_PAIRS + 3) / 4, KSPLIT);    // 3832 waves
    k_gemm_gates<<<ggrd, gblk, 0, stream>>>(x_bf, h_l_bf, wih_bf, whh_bf, gpart);

    k_reduce_gates<<<(unsigned)((ngelem + 255) / 256), 256, 0, stream>>>(
        gpart, b_ih, b_hh, gates);

    k_lstm<<<(BATCH_B * F_DIM + 255) / 256, 256, 0, stream>>>(gates, c_l, h_l, h_l_bf);

    k_dct_fwd<<<(BATCH_B * I_DIM + 255) / 256, 256, 0, stream>>>(Din, x_t, u_ih, I_DIM);
    k_dct_fwd<<<(BATCH_B * H_DIM + 255) / 256, 256, 0, stream>>>(Dhid, h, u_hh, H_DIM);

    k_sparse<<<BATCH_B, 256, 0, stream>>>(h_l, scaler, ih_r, ih_c, c_ih,
                                          hh_r, hh_c, c_hh, u_ih, u_hh, v);

    k_final<<<(BATCH_B * H_DIM + 255) / 256, 256, 0, stream>>>(
        Dhid, v, bias, h, (float*)d_out + (size_t)t * BATCH_B * H_DIM);
  }
}
